// JambaMambaBlock_65687229825530
// MI455X (gfx1250) — compile-verified, matrix operand order checked by reading, not run on hardware
//
#include <hip/hip_runtime.h>
#include <stdint.h>

// ---------------------------------------------------------------------------
// JambaMambaBlock for MI455X (gfx1250)
// bf16 WMMA GEMMs, double-buffered LDS tiles filled with async global->LDS
// copies (ASYNCcnt path), wave32 selective scan.
// ---------------------------------------------------------------------------

typedef __attribute__((ext_vector_type(16))) __bf16 v16bf;
typedef __attribute__((ext_vector_type(8)))  float  v8f;

#define D_MODEL 2048
#define D_STATE 16
#define D_CONV  4
#define B_SZ    2
#define T_LEN   2048
#define M_ROWS  (B_SZ * T_LEN)              // 4096
#define N_PROJ  (D_MODEL + 2 * D_STATE)     // 2080
#define NP_PAD  2176                        // 17*128, padded for 128-wide tiles

#define LDS_STRIDE 40                       // 32 data + 8 pad bf16 (80B rows)
#define TILE_ELEMS (128 * LDS_STRIDE)
#define TILE_BYTES (TILE_ELEMS * 2)         // 10240 B per tile

__device__ __forceinline__ uint16_t f32_to_bf16(float f) {
    union { float f; uint32_t u; } v; v.f = f;
    uint32_t u = v.u;
    uint32_t r = u + 0x7FFFu + ((u >> 16) & 1u);   // round-to-nearest-even
    if ((u & 0x7F800000u) == 0x7F800000u) r = u;   // inf/nan: truncate
    return (uint16_t)(r >> 16);
}

// ---------------------------- RMSNorm -> bf16 ------------------------------
__global__ __launch_bounds__(256) void rmsnorm_bf16_kernel(
    const float* __restrict__ x, const float* __restrict__ w,
    uint16_t* __restrict__ xn)
{
    __shared__ float red[256];
    int row = blockIdx.x;
    const float* xr = x + (long)row * D_MODEL;
    float s = 0.f;
    for (int c = threadIdx.x; c < D_MODEL; c += 256) { float v = xr[c]; s += v * v; }
    red[threadIdx.x] = s;
    __syncthreads();
    for (int st = 128; st > 0; st >>= 1) {
        if ((int)threadIdx.x < st) red[threadIdx.x] += red[threadIdx.x + st];
        __syncthreads();
    }
    float scale = rsqrtf(red[0] / (float)D_MODEL + 1e-6f);
    uint16_t* o = xn + (long)row * D_MODEL;
    for (int c = threadIdx.x; c < D_MODEL; c += 256)
        o[c] = f32_to_bf16(xr[c] * scale * w[c]);
}

// ------------------- f32 -> bf16 with optional zero-pad --------------------
__global__ __launch_bounds__(256) void cvt_bf16_kernel(
    const float* __restrict__ in, uint16_t* __restrict__ out,
    long n_src, long n_total)
{
    long i = (long)blockIdx.x * blockDim.x + threadIdx.x;
    if (i >= n_total) return;
    out[i] = (i < n_src) ? f32_to_bf16(in[i]) : (uint16_t)0;
}

// ------------------- WMMA bf16 GEMM: C = A * W^T (+resid) ------------------
// A: M x K bf16, W: N x K bf16 (row-major). Block tile 128x128, k-chunk 32,
// double-buffered through LDS via GLOBAL_LOAD_ASYNC_TO_LDS_B128 (ASYNCcnt).
// 8 waves (2x4); each wave computes 64x32.
// M % 128 == 0, N % 128 == 0, K % 32 == 0 required (arranged by caller).
__global__ __launch_bounds__(256) void wmma_gemm_lds(
    const uint16_t* __restrict__ A, const uint16_t* __restrict__ W,
    float* __restrict__ C, const float* __restrict__ resid,
    int M, int N, int K)
{
    __shared__ __align__(16) uint16_t lds[4 * TILE_ELEMS];  // A0,A1,B0,B1 = 40KB
    (void)M;

    const int tid  = threadIdx.x;
    const int wave = tid >> 5;
    const int lane = tid & 31;
    const int r    = lane & 15;            // row-in-tile / output column
    const int half = lane >> 4;            // K-half selector
    const int wm   = wave >> 2;            // 0..1 -> M offset wm*64
    const int wn   = wave & 3;             // 0..3 -> N offset wn*32
    const int m0   = blockIdx.y * 128;
    const int n0   = blockIdx.x * 128;

    // wave-relative LDS byte offset of the tile buffers (low 32 bits of the
    // generic address of a __shared__ object are the LDS offset)
    const uint32_t lds_base = (uint32_t)(uintptr_t)&lds[0];

    // stage one 128x32 A tile + 128x32 B tile into LDS buffer `buf` using the
    // CDNA5 async global->LDS copy path (no VGPR roundtrip; ASYNCcnt-tracked)
    auto stage_async = [&](int k0, int buf) {
        uint32_t Al = lds_base + (buf ? TILE_BYTES : 0);
        uint32_t Bl = lds_base + 2 * TILE_BYTES + (buf ? TILE_BYTES : 0);
#pragma unroll
        for (int rnd = 0; rnd < 2; ++rnd) {
            int chunk = tid + rnd * 256;       // 0..511 : 128 rows x 4 x 8 elems
            int row = chunk >> 2;
            int c4  = chunk & 3;
            uint64_t ga = (uint64_t)(uintptr_t)(A + (long)(m0 + row) * K + k0 + c4 * 8);
            uint64_t gb = (uint64_t)(uintptr_t)(W + (long)(n0 + row) * K + k0 + c4 * 8);
            uint32_t la = Al + (uint32_t)(row * (LDS_STRIDE * 2) + c4 * 16);
            uint32_t lb = Bl + (uint32_t)(row * (LDS_STRIDE * 2) + c4 * 16);
            asm volatile("global_load_async_to_lds_b128 %0, %1, off"
                         :: "v"(la), "v"(ga) : "memory");
            asm volatile("global_load_async_to_lds_b128 %0, %1, off"
                         :: "v"(lb), "v"(gb) : "memory");
        }
    };

    v8f acc[4][2] = {};

    stage_async(0, 0);
    asm volatile("s_wait_asynccnt 0x0" ::: "memory");
    __syncthreads();

    const int KT = K >> 5;
    for (int kc = 0; kc < KT; ++kc) {
        int cur = kc & 1;
        if (kc + 1 < KT) stage_async((kc + 1) << 5, cur ^ 1);

        const uint16_t* Ab = lds + (cur ? TILE_ELEMS : 0);
        const uint16_t* Bb = lds + (cur ? 3 * TILE_ELEMS : 2 * TILE_ELEMS);

        union Frag { uint4 q[2]; v16bf v; };
        Frag af[4], bfr[2];
#pragma unroll
        for (int i = 0; i < 4; ++i) {
            const uint16_t* p = Ab + (wm * 64 + i * 16 + r) * LDS_STRIDE + half * 16;
            af[i].q[0] = *(const uint4*)p;
            af[i].q[1] = *(const uint4*)(p + 8);
        }
#pragma unroll
        for (int j = 0; j < 2; ++j) {
            const uint16_t* p = Bb + (wn * 32 + j * 16 + r) * LDS_STRIDE + half * 16;
            bfr[j].q[0] = *(const uint4*)p;
            bfr[j].q[1] = *(const uint4*)(p + 8);
        }
#pragma unroll
        for (int i = 0; i < 4; ++i)
#pragma unroll
            for (int j = 0; j < 2; ++j)
                acc[i][j] = __builtin_amdgcn_wmma_f32_16x16x32_bf16(
                    false, af[i].v, false, bfr[j].v, (short)0, acc[i][j], false, false);

        // drain this wave's async copies into the next buffer, then block-sync
        asm volatile("s_wait_asynccnt 0x0" ::: "memory");
        __syncthreads();
    }

    // 16x16 f32 C/D layout: VGPR e -> M = half*8 + e, N = lane&15
#pragma unroll
    for (int i = 0; i < 4; ++i) {
        int mb = m0 + wm * 64 + i * 16 + half * 8;
#pragma unroll
        for (int j = 0; j < 2; ++j) {
            int nb = n0 + wn * 32 + j * 16 + r;
            if (resid) {
#pragma unroll
                for (int e = 0; e < 8; ++e) {
                    long idx = (long)(mb + e) * N + nb;
                    C[idx] = acc[i][j][e] + resid[idx];
                }
            } else {
#pragma unroll
                for (int e = 0; e < 8; ++e) {
                    long idx = (long)(mb + e) * N + nb;
                    C[idx] = acc[i][j][e];
                }
            }
        }
    }
}

// ----------------- depthwise causal conv(4) + SiLU -> f32+bf16 -------------
__global__ __launch_bounds__(256) void conv_silu_kernel(
    const float* __restrict__ xz, const float* __restrict__ cw,
    const float* __restrict__ cb, float* __restrict__ xssm,
    uint16_t* __restrict__ xssm_bf)
{
    long idx = (long)blockIdx.x * blockDim.x + threadIdx.x;
    long total = (long)M_ROWS * D_MODEL;
    if (idx >= total) return;
    int  d = (int)(idx % D_MODEL);
    long m = idx / D_MODEL;               // m = b*T + t
    int  t = (int)(m % T_LEN);
    float acc = cb[d];
#pragma unroll
    for (int j = 0; j < D_CONV; ++j) {
        int tt = t - (D_CONV - 1) + j;
        if (tt >= 0)
            acc += cw[d * D_CONV + j] * xz[(m + (long)(tt - t)) * (2 * D_MODEL) + d];
    }
    float s = acc / (1.f + __expf(-acc)); // silu
    xssm[idx]    = s;
    xssm_bf[idx] = f32_to_bf16(s);
}

// ------------------------------ selective scan -----------------------------
// 16 lanes per (b,d) channel; lane = state index n. Sequential over T.
__global__ __launch_bounds__(256) void scan_kernel(
    const float* __restrict__ proj,   // M x NP_PAD (only first 2080 cols valid)
    const float* __restrict__ xssm,   // M x 2048
    const float* __restrict__ A_log,  // D x 16
    const float* __restrict__ Dw,     // D
    const float* __restrict__ h_in,   // B x D x 16
    float* __restrict__ y,            // M x 2048
    float* __restrict__ h_out)        // B x D x 16
{
    int grp = (int)((blockIdx.x * blockDim.x + threadIdx.x) >> 4);
    int n   = threadIdx.x & 15;
    if (grp >= B_SZ * D_MODEL) return;
    int b = grp / D_MODEL;
    int d = grp % D_MODEL;

    float A  = -__expf(A_log[d * D_STATE + n]);
    float h  = h_in[(long)grp * D_STATE + n];
    float dc = Dw[d];
    long base = (long)b * T_LEN;

    for (int t = 0; t < T_LEN; ++t) {
        long row = base + t;
        const float* pr = proj + row * NP_PAD;
        float draw  = pr[d];
        float delta = (draw > 20.f) ? draw : log1pf(__expf(draw));
        float xt    = xssm[row * D_MODEL + d];
        float Bn    = pr[D_MODEL + n];
        float Cn    = pr[D_MODEL + D_STATE + n];
        h = __expf(delta * A) * h + (delta * xt) * Bn;
        float v = h * Cn;
        v += __shfl_xor(v, 1, 16);
        v += __shfl_xor(v, 2, 16);
        v += __shfl_xor(v, 4, 16);
        v += __shfl_xor(v, 8, 16);
        if (n == 0) y[row * D_MODEL + d] = v + dc * xt;
    }
    h_out[(long)grp * D_STATE + n] = h;
}

// ---------------------------- y * silu(z) -> bf16 --------------------------
__global__ __launch_bounds__(256) void gate_bf16_kernel(
    const float* __restrict__ y, const float* __restrict__ xz,
    uint16_t* __restrict__ out_bf)
{
    long idx = (long)blockIdx.x * blockDim.x + threadIdx.x;
    long total = (long)M_ROWS * D_MODEL;
    if (idx >= total) return;
    int  d = (int)(idx % D_MODEL);
    long m = idx / D_MODEL;
    float z = xz[m * (2 * D_MODEL) + D_MODEL + d];
    float g = z / (1.f + __expf(-z));
    out_bf[idx] = f32_to_bf16(y[idx] * g);
}

// ---------------------------------------------------------------------------
extern "C" void kernel_launch(void* const* d_in, const int* in_sizes, int n_in,
                              void* d_out, int out_size, void* d_ws, size_t ws_size,
                              hipStream_t stream) {
    (void)in_sizes; (void)n_in; (void)out_size; (void)ws_size;
    const float* x        = (const float*)d_in[0];   // B,T,D
    const float* h_in     = (const float*)d_in[1];   // B,D,N
    const float* norm_w   = (const float*)d_in[2];   // D
    const float* in_w     = (const float*)d_in[3];   // 2D,D
    const float* conv_w   = (const float*)d_in[4];   // D,1,4
    const float* conv_b   = (const float*)d_in[5];   // D
    const float* xproj_w  = (const float*)d_in[6];   // 2080,D
    const float* A_log    = (const float*)d_in[7];   // D,16
    const float* Dw       = (const float*)d_in[8];   // D
    const float* out_w    = (const float*)d_in[9];   // D,D

    float* out   = (float*)d_out;                              // M x D
    float* h_out = (float*)d_out + (long)M_ROWS * D_MODEL;     // B x D x N

    // workspace layout (bytes); total ~194 MB
    char* ws = (char*)d_ws;
    const long OFF_W  = 16777216;                  // A_bf is 16 MB at offset 0
    const long OFF_XZ = 33554432;                  // 64 MB (M x 4096 f32)
    const long OFF_XS = OFF_XZ + 67108864L;        // 32 MB (M x 2048 f32)
    const long OFF_PR = OFF_XS + 33554432L;        // ~34 MB (M x 2176 f32)
    const long OFF_Y  = OFF_PR + (long)M_ROWS * NP_PAD * 4;   // 32 MB

    uint16_t* A_bf = (uint16_t*)(ws);
    uint16_t* W_bf = (uint16_t*)(ws + OFF_W);
    float*    xz   = (float*)(ws + OFF_XZ);
    float*    xssm = (float*)(ws + OFF_XS);
    float*    proj = (float*)(ws + OFF_PR);
    float*    ybuf = (float*)(ws + OFF_Y);

    const long elems = (long)M_ROWS * D_MODEL;     // 8388608
    const int  TB = 256;
    auto nblk = [](long n, int tb) { return (int)((n + tb - 1) / tb); };

    // 1) RMSNorm -> bf16 activations
    rmsnorm_bf16_kernel<<<M_ROWS, TB, 0, stream>>>(x, norm_w, A_bf);

    // 2) in_proj GEMM: xz = xn @ in_w^T   (N=4096)
    {
        long nw = (long)2 * D_MODEL * D_MODEL;
        cvt_bf16_kernel<<<nblk(nw, TB), TB, 0, stream>>>(in_w, W_bf, nw, nw);
        dim3 grid((2 * D_MODEL) / 128, M_ROWS / 128);
        wmma_gemm_lds<<<grid, TB, 0, stream>>>(A_bf, W_bf, xz, nullptr,
                                               M_ROWS, 2 * D_MODEL, D_MODEL);
    }

    // 3) depthwise conv + silu -> x_ssm (f32 + bf16 into A_bf)
    conv_silu_kernel<<<nblk(elems, TB), TB, 0, stream>>>(xz, conv_w, conv_b, xssm, A_bf);

    // 4) x_proj GEMM: proj = x_ssm @ xproj_w^T   (N=2080 zero-padded to 2176)
    {
        long ns = (long)N_PROJ * D_MODEL;
        long nt = (long)NP_PAD * D_MODEL;
        cvt_bf16_kernel<<<nblk(nt, TB), TB, 0, stream>>>(xproj_w, W_bf, ns, nt);
        dim3 grid(NP_PAD / 128, M_ROWS / 128);
        wmma_gemm_lds<<<grid, TB, 0, stream>>>(A_bf, W_bf, proj, nullptr,
                                               M_ROWS, NP_PAD, D_MODEL);
    }

    // 5) selective scan (sequential over T, 16 lanes per channel)
    scan_kernel<<<nblk((long)B_SZ * D_MODEL * 16, TB), TB, 0, stream>>>(
        proj, xssm, A_log, Dw, h_in, ybuf, h_out);

    // 6) gate: y * silu(z) -> bf16 into A_bf
    gate_bf16_kernel<<<nblk(elems, TB), TB, 0, stream>>>(ybuf, xz, A_bf);

    // 7) out_proj GEMM + residual: out = x + yg @ out_w^T   (N=2048)
    {
        long nw = (long)D_MODEL * D_MODEL;
        cvt_bf16_kernel<<<nblk(nw, TB), TB, 0, stream>>>(out_w, W_bf, nw, nw);
        dim3 grid(D_MODEL / 128, M_ROWS / 128);
        wmma_gemm_lds<<<grid, TB, 0, stream>>>(A_bf, W_bf, out, x,
                                               M_ROWS, D_MODEL, D_MODEL);
    }
}
